// HGTFeatureExtractor_42442866819874
// MI455X (gfx1250) — matrix operand order repaired and checked by reading, not compiled, not run on hardware
//
#include <hip/hip_runtime.h>

// ---------------- problem constants (match reference) ----------------
#define Bg   8
#define Nn   1024
#define Mm   64
#define OPFc 16
#define MFc  8
#define Hh   8
#define Dd   64
#define Lc   3
#define Ff   512          // H*D
#define KCc  8
#define NOPc (Bg*Nn)      // 8192
#define NMc  (Bg*Mm)      // 512
#define E1c  (NOPc*KCc)   // 65536 canproc edges (machine->op)
#define E2c  (NOPc*KCc)   // 65536 compat edges  (op->machine)
#define EPSc 1e-5f

typedef __attribute__((ext_vector_type(16))) __bf16 v16bf;
typedef __attribute__((ext_vector_type(8)))  __bf16 v8bf;
typedef __attribute__((ext_vector_type(4)))  __bf16 v4bf;
typedef __attribute__((ext_vector_type(8)))  float  v8f;

// ==================== generic bf16-WMMA GEMM ====================
// C[M x Ncols] = X[M x K] @ W[K x Ncols] (+ bias), fp32 in/out, bf16 WMMA math.
// REQUIRES: Mrows % 128 == 0, Ncols % 64 == 0, K % 4 == 0 (all call sites comply).
// FULLK=true when K % 32 == 0 -> branch-free staging (all heavy GEMMs).
// Batched (per-head rel transforms) via element strides xbs/wbs/cbs.
#define BM 128
#define BN 64
#define BK 32

template <bool FULLK>
__global__ __launch_bounds__(256)
void gemm_bf16_wmma(const float* __restrict__ Xp, int ldx, long long xbs,
                    const float* __restrict__ Wp, int ldw, long long wbs,
                    const float* __restrict__ bias,
                    float* __restrict__ Cp, int ldc, long long cbs,
                    int Kdim)
{
    const float* X = Xp + (long long)blockIdx.z * xbs;
    const float* W = Wp + (long long)blockIdx.z * wbs;
    float*       C = Cp + (long long)blockIdx.z * cbs;

    const int rowBase = blockIdx.x * BM;
    const int colBase = blockIdx.y * BN;

    __shared__ __bf16 lA [BM][BK + 8];   // k-major rows (A fragment reads contiguous)
    __shared__ __bf16 lBt[BN][BK + 8];   // TRANSPOSED: [col][k] (B fragment reads contiguous)

    const int tid  = threadIdx.x;
    const int lane = tid & 31;
    const int wid  = tid >> 5;     // 0..7
    const int wr   = wid & 3;      // wave row slab (32 rows)
    const int wc   = wid >> 2;     // wave col slab (32 cols)
    const int hl   = lane >> 4;    // lane half
    const int lrow = lane & 15;

    v8f acc[2][2] = {};

    const int nk = (Kdim + BK - 1) / BK;
    for (int kt = 0; kt < nk; ++kt) {
        const int k0g = kt * BK;

        // ---- phase 1: issue ALL global loads (6 x b128 in flight per thread) ----
        float4 va[4], vb[2];
        #pragma unroll
        for (int it = 0; it < 4; ++it) {
            int i  = tid + it * 256;           // 0..1023 quads (A: 128x32)
            int r  = i >> 3;                   // 0..127
            int cq = (i & 7) << 2;             // 0,4,..,28
            int gc = k0g + cq;
            va[it] = make_float4(0.f, 0.f, 0.f, 0.f);
            if (FULLK || gc + 4 <= Kdim)
                va[it] = *(const float4*)(X + (long long)(rowBase + r) * ldx + gc);
        }
        #pragma unroll
        for (int it = 0; it < 2; ++it) {
            int i  = tid + it * 256;           // 0..511 quads (B: 32x64)
            int k  = i >> 4;                   // 0..31
            int cg = (i & 15) << 2;            // 0,4,..,60
            vb[it] = make_float4(0.f, 0.f, 0.f, 0.f);
            if (FULLK || k0g + k < Kdim)
                vb[it] = *(const float4*)(W + (long long)(k0g + k) * ldw + colBase + cg);
        }
        // prefetch next k-tile while this one is consumed
        if (kt + 1 < nk) {
            __builtin_prefetch(X + (long long)(rowBase + (tid & 127)) * ldx + k0g + BK, 0, 0);
            if (tid < BK)
                __builtin_prefetch(W + (long long)(k0g + BK + tid) * ldw + colBase, 0, 0);
        }

        // ---- phase 2: convert fp32 -> bf16 and stage into LDS ----
        #pragma unroll
        for (int it = 0; it < 4; ++it) {
            int i  = tid + it * 256;
            int r  = i >> 3;
            int cq = (i & 7) << 2;
            v4bf b; b[0] = (__bf16)va[it].x; b[1] = (__bf16)va[it].y;
                    b[2] = (__bf16)va[it].z; b[3] = (__bf16)va[it].w;
            *(v4bf*)&lA[r][cq] = b;
        }
        #pragma unroll
        for (int it = 0; it < 2; ++it) {
            int i  = tid + it * 256;
            int k  = i >> 4;
            int cg = (i & 15) << 2;
            lBt[cg + 0][k] = (__bf16)vb[it].x;
            lBt[cg + 1][k] = (__bf16)vb[it].y;
            lBt[cg + 2][k] = (__bf16)vb[it].z;
            lBt[cg + 3][k] = (__bf16)vb[it].w;
        }
        __syncthreads();

        // ---- fragments (all contiguous 128-bit LDS reads) ----
        v16bf afr[2], bfr[2];
        #pragma unroll
        for (int i = 0; i < 2; ++i) {
            int rb = wr * 32 + i * 16 + lrow;
            v8bf lo = *(const v8bf*)&lA[rb][hl * 8];        // K = hl*8 .. +7
            v8bf hi = *(const v8bf*)&lA[rb][16 + hl * 8];   // K = 16+hl*8 .. +7
            #pragma unroll
            for (int e = 0; e < 8; ++e) { afr[i][e] = lo[e]; afr[i][8 + e] = hi[e]; }
        }
        #pragma unroll
        for (int j = 0; j < 2; ++j) {
            int cb = wc * 32 + j * 16 + lrow;
            v8bf lo = *(const v8bf*)&lBt[cb][hl * 8];       // K = hl*8 .. +7
            v8bf hi = *(const v8bf*)&lBt[cb][16 + hl * 8];  // K = 16+hl*8 .. +7
            #pragma unroll
            for (int e = 0; e < 8; ++e) { bfr[j][e] = lo[e]; bfr[j][8 + e] = hi[e]; }
        }
        #pragma unroll
        for (int i = 0; i < 2; ++i)
            #pragma unroll
            for (int j = 0; j < 2; ++j)
                acc[i][j] = __builtin_amdgcn_wmma_f32_16x16x32_bf16(
                    false, afr[i], false, bfr[j], (short)0, acc[i][j], false, false);
        __syncthreads();
    }

    // epilogue: C/D layout -> row = hl*8 + r, col = lrow (rows always in-bounds)
    #pragma unroll
    for (int i = 0; i < 2; ++i)
        #pragma unroll
        for (int j = 0; j < 2; ++j) {
            int gc = colBase + wc * 32 + j * 16 + lrow;
            float bv = bias ? bias[gc] : 0.0f;
            #pragma unroll
            for (int r = 0; r < 8; ++r) {
                int gr = rowBase + wr * 32 + i * 16 + hl * 8 + r;
                C[(long long)gr * ldc + gc] = acc[i][j][r] + bv;
            }
        }
}

// ==================== elementwise / reduction kernels ====================
__global__ void fill_kernel(float* p, long long n, float v) {
    long long i = (long long)blockIdx.x * blockDim.x + threadIdx.x;
    if (i < n) p[i] = v;
}

__global__ void copy_kernel(const float* __restrict__ s, float* __restrict__ d, long long n) {
    long long i = (long long)blockIdx.x * blockDim.x + threadIdx.x;
    if (i < n) d[i] = s[i];
}

__global__ void gelu_kernel(float* p, long long n) {
    long long i = (long long)blockIdx.x * blockDim.x + threadIdx.x;
    if (i < n) { float v = p[i]; p[i] = 0.5f * v * (1.0f + erff(v * 0.70710678118654752f)); }
}

__global__ void combine_kernel(const float* __restrict__ o, const float* __restrict__ x,
                               const float* __restrict__ sk, float* __restrict__ out, long long n) {
    long long i = (long long)blockIdx.x * blockDim.x + threadIdx.x;
    if (i < n) {
        float g = 1.0f / (1.0f + expf(-sk[0]));
        out[i] = g * o[i] + (1.0f - g) * x[i] + x[i];
    }
}

__global__ __launch_bounds__(256)
void stats_kernel(const float* __restrict__ x, long long n, float* __restrict__ st) {
    __shared__ float s1[256], s2[256];
    long long i0 = (long long)blockIdx.x * blockDim.x + threadIdx.x;
    long long stride = (long long)gridDim.x * blockDim.x;
    float a = 0.0f, b = 0.0f;
    for (long long i = i0; i < n; i += stride) { float v = x[i]; a += v; b += v * v; }
    s1[threadIdx.x] = a; s2[threadIdx.x] = b;
    __syncthreads();
    for (int w = 128; w > 0; w >>= 1) {
        if ((int)threadIdx.x < w) { s1[threadIdx.x] += s1[threadIdx.x + w]; s2[threadIdx.x] += s2[threadIdx.x + w]; }
        __syncthreads();
    }
    if (threadIdx.x == 0) { atomicAdd(&st[0], s1[0]); atomicAdd(&st[1], s2[0]); }
}

// graph LayerNorm: scalar mean/std over the whole tensor, per-column affine
__global__ void apply_ln_kernel(const float* __restrict__ xin, float* __restrict__ xout,
                                long long n, const float* __restrict__ w, const float* __restrict__ b,
                                const float* __restrict__ st, float invn) {
    long long i = (long long)blockIdx.x * blockDim.x + threadIdx.x;
    if (i >= n) return;
    float mean = st[0] * invn;
    float var  = fmaxf(st[1] * invn - mean * mean, 0.0f);
    float inv  = 1.0f / (sqrtf(var) + EPSc);
    int col = (int)(i % Ff);
    xout[i] = (xin[i] - mean) * inv * w[col] + b[col];
}

// ==================== op-node attention (structured, no atomics) ====================
// Each op node d has <=1 precedes in-edge (src=d-1 iff d%N!=0) and exactly 8
// canproc in-edges (sources at canproc_src[d*8 .. d*8+7]). Joint softmax per head.
__global__ __launch_bounds__(256)
void op_attention_kernel(const float* __restrict__ QO,
                         const float* __restrict__ K0, const float* __restrict__ V0,
                         const float* __restrict__ K1, const float* __restrict__ V1,
                         const int* __restrict__ canproc_src,
                         const float* __restrict__ p0, const float* __restrict__ p1,
                         float* __restrict__ AGG) {
    const int d = blockIdx.x;
    __shared__ float sc[9][Hh];
    __shared__ float wgt[9][Hh];
    __shared__ int   srcs[9];
    const int tid = threadIdx.x;
    const bool hasPre = (d % Nn) != 0;
    if (tid == 0) srcs[0] = hasPre ? d - 1 : -1;
    if (tid >= 1 && tid < 9) srcs[tid] = canproc_src[d * KCc + (tid - 1)];
    __syncthreads();

    if (tid < 72) {
        int j = tid / Hh, h = tid % Hh;
        float a = -3.0e38f;
        int s = srcs[j];
        if (s >= 0) {
            const float* q = QO + (long long)d * Ff + h * Dd;
            const float* k = (j == 0 ? K0 : K1) + (long long)s * Ff + h * Dd;
            float dot = 0.0f;
            #pragma unroll 8
            for (int e = 0; e < Dd; ++e) dot += q[e] * k[e];
            a = dot * (j == 0 ? p0[h] : p1[h]) * 0.125f; // 1/sqrt(64)
        }
        sc[j][h] = a;
    }
    __syncthreads();
    if (tid < Hh) {
        int h = tid;
        float mx = -3.0e38f;
        for (int j = 0; j < 9; ++j) mx = fmaxf(mx, sc[j][h]);
        float ssum = 0.0f;
        for (int j = 0; j < 9; ++j) {
            float e = (sc[j][h] <= -1.0e38f) ? 0.0f : expf(sc[j][h] - mx);
            wgt[j][h] = e; ssum += e;
        }
        float inv = 1.0f / (ssum + 1e-16f);
        for (int j = 0; j < 9; ++j) wgt[j][h] *= inv;
    }
    __syncthreads();
    for (int f = tid; f < Ff; f += 256) {
        int h = f >> 6;
        float acc = 0.0f;
        #pragma unroll
        for (int j = 0; j < 9; ++j) {
            int s = srcs[j];
            if (s < 0) continue;
            const float* v = (j == 0 ? V0 : V1) + (long long)s * Ff;
            acc += wgt[j][h] * v[f];
        }
        AGG[(long long)d * Ff + f] = acc;
    }
}

// ==================== machine-node attention (atomic segment softmax) ==========
__device__ inline void atomicMaxF(float* addr, float v) {
    if (v >= 0.0f) atomicMax((int*)addr, __float_as_int(v));
    else           atomicMin((unsigned int*)addr, __float_as_uint(v));
}

__global__ void edge_score2_kernel(const float* __restrict__ QM, const float* __restrict__ K2,
                                   const int* __restrict__ s2, const int* __restrict__ d2,
                                   const float* __restrict__ p2,
                                   float* __restrict__ att, float* __restrict__ smax) {
    long long i = (long long)blockIdx.x * blockDim.x + threadIdx.x;
    if (i >= (long long)E2c * Hh) return;
    int e = (int)(i / Hh), h = (int)(i % Hh);
    int s = s2[e], d = d2[e];
    const float* q = QM + (long long)d * Ff + h * Dd;
    const float* k = K2 + (long long)s * Ff + h * Dd;
    float dot = 0.0f;
    #pragma unroll 8
    for (int t = 0; t < Dd; ++t) dot += q[t] * k[t];
    float a = dot * p2[h] * 0.125f;
    att[i] = a;
    atomicMaxF(&smax[d * Hh + h], a);
}

__global__ void edge_exp2_kernel(float* __restrict__ att, const int* __restrict__ d2,
                                 const float* __restrict__ smax, float* __restrict__ ssum) {
    long long i = (long long)blockIdx.x * blockDim.x + threadIdx.x;
    if (i >= (long long)E2c * Hh) return;
    int e = (int)(i / Hh), h = (int)(i % Hh);
    int d = d2[e];
    float ex = expf(att[i] - smax[d * Hh + h]);
    att[i] = ex;
    atomicAdd(&ssum[d * Hh + h], ex);
}

__global__ __launch_bounds__(256)
void edge_agg2_kernel(const float* __restrict__ att, const int* __restrict__ s2,
                      const int* __restrict__ d2, const float* __restrict__ ssum,
                      const float* __restrict__ V2, float* __restrict__ AGGM) {
    const int e = blockIdx.x;
    const int s = s2[e], d = d2[e];
    for (int f = threadIdx.x; f < Ff; f += 256) {
        int h = f >> 6;
        float w = att[(long long)e * Hh + h] / (ssum[d * Hh + h] + 1e-16f);
        atomicAdd(&AGGM[(long long)d * Ff + f], V2[(long long)s * Ff + f] * w);
    }
}

// ==================== per-graph feature means ====================
__global__ void mean_rows_kernel(const float* __restrict__ x, float* __restrict__ out, int rowsPerB) {
    int i = blockIdx.x * blockDim.x + threadIdx.x;   // over Bg*Ff
    if (i >= Bg * Ff) return;
    int b = i / Ff, f = i % Ff;
    float s = 0.0f;
    for (int r = 0; r < rowsPerB; ++r) s += x[((long long)b * rowsPerB + r) * Ff + f];
    out[i] = s / (float)rowsPerB;
}

// ==================== host orchestration ====================
static inline void gemm(hipStream_t st, const float* X, int ldx, long long xbs,
                        const float* W, int ldw, long long wbs, const float* bias,
                        float* C, int ldc, long long cbs, int Mr, int Nc, int K, int batches) {
    dim3 grid((Mr + BM - 1) / BM, Nc / BN, batches);
    if ((K % BK) == 0)
        gemm_bf16_wmma<true ><<<grid, 256, 0, st>>>(X, ldx, xbs, W, ldw, wbs, bias, C, ldc, cbs, K);
    else
        gemm_bf16_wmma<false><<<grid, 256, 0, st>>>(X, ldx, xbs, W, ldw, wbs, bias, C, ldc, cbs, K);
}
static inline dim3 g1(long long n) { return dim3((unsigned)((n + 255) / 256)); }

extern "C" void kernel_launch(void* const* d_in, const int* in_sizes, int n_in,
                              void* d_out_v, int out_size, void* d_ws, size_t ws_size,
                              hipStream_t stream) {
    (void)in_sizes; (void)n_in; (void)out_size; (void)ws_size;
    const float* op_x      = (const float*)d_in[0];
    const float* machine_x = (const float*)d_in[1];
    const float* W_emb_op  = (const float*)d_in[2];
    const float* b_emb_op  = (const float*)d_in[3];
    const float* W_emb_m   = (const float*)d_in[4];
    const float* b_emb_m   = (const float*)d_in[5];
    const float* opn_w     = (const float*)d_in[6];
    const float* opn_b     = (const float*)d_in[7];
    const float* mn_w      = (const float*)d_in[8];
    const float* mn_b      = (const float*)d_in[9];
    const float* Wk        = (const float*)d_in[10];
    const float* bk        = (const float*)d_in[11];
    const float* Wq        = (const float*)d_in[12];
    const float* bq        = (const float*)d_in[13];
    const float* Wv        = (const float*)d_in[14];
    const float* bv        = (const float*)d_in[15];
    const float* A_rel     = (const float*)d_in[16];
    const float* M_rel     = (const float*)d_in[17];
    const float* p_rel     = (const float*)d_in[18];
    const float* W_out     = (const float*)d_in[19];
    const float* b_out     = (const float*)d_in[20];
    const float* skip      = (const float*)d_in[21];
    const float* ln_w      = (const float*)d_in[22];
    const float* ln_b      = (const float*)d_in[23];
    const int*   ei_canproc= (const int*)d_in[25];   // [src(E1) | dst(E1)]
    const int*   ei_compat = (const int*)d_in[26];   // [src(E2) | dst(E2)]
    float* d_out = (float*)d_out_v;

    // ---- workspace carve (floats) ----
    const size_t OSZ = (size_t)NOPc * Ff;     // 4,194,304
    const size_t MSZ = (size_t)NMc  * Ff;     //   262,144
    float* w = (float*)d_ws;
    float* X_OP = w;  w += OSZ;
    float* QO   = w;  w += OSZ;
    float* KO   = w;  w += OSZ;   // also holds op output-proj result
    float* VO   = w;  w += OSZ;   // also combine temp
    float* K0   = w;  w += OSZ;
    float* V0   = w;  w += OSZ;
    float* K2   = w;  w += OSZ;
    float* V2   = w;  w += OSZ;
    float* AGGO = w;  w += OSZ;
    float* X_M  = w;  w += MSZ;
    float* QM   = w;  w += MSZ;
    float* KM   = w;  w += MSZ;   // also machine output-proj result
    float* VM   = w;  w += MSZ;   // also combine temp
    float* K1   = w;  w += MSZ;
    float* V1   = w;  w += MSZ;
    float* AGGM = w;  w += MSZ;
    float* ATT2 = w;  w += (size_t)E2c * Hh;
    float* SMAX = w;  w += (size_t)NMc * Hh;
    float* SSUM = w;  w += (size_t)NMc * Hh;
    float* ST   = w;  w += 4;

    const int* cp_src = ei_canproc;            // machine sources, grouped 8 per op
    const int* c2_src = ei_compat;             // op sources
    const int* c2_dst = ei_compat + E2c;       // machine dests

    // ---- embeddings + graph LN ----
    gemm(stream, op_x, OPFc, 0, W_emb_op, Ff, 0, b_emb_op, VO, Ff, 0, NOPc, Ff, OPFc, 1);
    fill_kernel<<<g1(2), 256, 0, stream>>>(ST, 2, 0.0f);
    stats_kernel<<<512, 256, 0, stream>>>(VO, (long long)OSZ, ST);
    apply_ln_kernel<<<g1(OSZ), 256, 0, stream>>>(VO, X_OP, (long long)OSZ, opn_w, opn_b, ST, 1.0f / (float)OSZ);

    gemm(stream, machine_x, MFc, 0, W_emb_m, Ff, 0, b_emb_m, VM, Ff, 0, NMc, Ff, MFc, 1);
    fill_kernel<<<g1(2), 256, 0, stream>>>(ST, 2, 0.0f);
    stats_kernel<<<128, 256, 0, stream>>>(VM, (long long)MSZ, ST);
    apply_ln_kernel<<<g1(MSZ), 256, 0, stream>>>(VM, X_M, (long long)MSZ, mn_w, mn_b, ST, 1.0f / (float)MSZ);

    // ---- HGT layers ----
    const long long FF = (long long)Ff * Ff;
    const long long HDD = (long long)Hh * Dd * Dd;
    for (int l = 0; l < Lc; ++l) {
        const float* Wq0 = Wq + (l * 2 + 0) * FF; const float* bq0 = bq + (l * 2 + 0) * Ff;
        const float* Wq1 = Wq + (l * 2 + 1) * FF; const float* bq1 = bq + (l * 2 + 1) * Ff;
        const float* Wk0 = Wk + (l * 2 + 0) * FF; const float* bk0 = bk + (l * 2 + 0) * Ff;
        const float* Wk1 = Wk + (l * 2 + 1) * FF; const float* bk1 = bk + (l * 2 + 1) * Ff;
        const float* Wv0 = Wv + (l * 2 + 0) * FF; const float* bv0 = bv + (l * 2 + 0) * Ff;
        const float* Wv1 = Wv + (l * 2 + 1) * FF; const float* bv1 = bv + (l * 2 + 1) * Ff;

        // K/Q/V projections (WMMA GEMMs)
        gemm(stream, X_OP, Ff, 0, Wq0, Ff, 0, bq0, QO, Ff, 0, NOPc, Ff, Ff, 1);
        gemm(stream, X_OP, Ff, 0, Wk0, Ff, 0, bk0, KO, Ff, 0, NOPc, Ff, Ff, 1);
        gemm(stream, X_OP, Ff, 0, Wv0, Ff, 0, bv0, VO, Ff, 0, NOPc, Ff, Ff, 1);
        gemm(stream, X_M,  Ff, 0, Wq1, Ff, 0, bq1, QM, Ff, 0, NMc,  Ff, Ff, 1);
        gemm(stream, X_M,  Ff, 0, Wk1, Ff, 0, bk1, KM, Ff, 0, NMc,  Ff, Ff, 1);
        gemm(stream, X_M,  Ff, 0, Wv1, Ff, 0, bv1, VM, Ff, 0, NMc,  Ff, Ff, 1);

        // per-head relation transforms (batched WMMA GEMMs over H)
        const float* A0 = A_rel + (l * 3 + 0) * HDD;  const float* M0 = M_rel + (l * 3 + 0) * HDD;
        const float* A1 = A_rel + (l * 3 + 1) * HDD;  const float* M1 = M_rel + (l * 3 + 1) * HDD;
        const float* A2 = A_rel + (l * 3 + 2) * HDD;  const float* M2 = M_rel + (l * 3 + 2) * HDD;
        gemm(stream, KO, Ff, Dd, A0, Dd, (long long)Dd * Dd, nullptr, K0, Ff, Dd, NOPc, Dd, Dd, Hh);
        gemm(stream, VO, Ff, Dd, M0, Dd, (long long)Dd * Dd, nullptr, V0, Ff, Dd, NOPc, Dd, Dd, Hh);
        gemm(stream, KM, Ff, Dd, A1, Dd, (long long)Dd * Dd, nullptr, K1, Ff, Dd, NMc,  Dd, Dd, Hh);
        gemm(stream, VM, Ff, Dd, M1, Dd, (long long)Dd * Dd, nullptr, V1, Ff, Dd, NMc,  Dd, Dd, Hh);
        gemm(stream, KO, Ff, Dd, A2, Dd, (long long)Dd * Dd, nullptr, K2, Ff, Dd, NOPc, Dd, Dd, Hh);
        gemm(stream, VO, Ff, Dd, M2, Dd, (long long)Dd * Dd, nullptr, V2, Ff, Dd, NOPc, Dd, Dd, Hh);

        // attention into op nodes (structured, joint softmax over 2 edge types)
        const float* p0 = p_rel + (l * 3 + 0) * Hh;
        const float* p1 = p_rel + (l * 3 + 1) * Hh;
        const float* p2 = p_rel + (l * 3 + 2) * Hh;
        op_attention_kernel<<<NOPc, 256, 0, stream>>>(QO, K0, V0, K1, V1, cp_src, p0, p1, AGGO);

        // attention into machine nodes (atomic segment softmax)
        fill_kernel<<<g1((long long)NMc * Hh), 256, 0, stream>>>(SMAX, (long long)NMc * Hh, -1.0e30f);
        fill_kernel<<<g1((long long)NMc * Hh), 256, 0, stream>>>(SSUM, (long long)NMc * Hh, 0.0f);
        fill_kernel<<<g1(MSZ), 256, 0, stream>>>(AGGM, (long long)MSZ, 0.0f);
        edge_score2_kernel<<<g1((long long)E2c * Hh), 256, 0, stream>>>(QM, K2, c2_src, c2_dst, p2, ATT2, SMAX);
        edge_exp2_kernel<<<g1((long long)E2c * Hh), 256, 0, stream>>>(ATT2, c2_dst, SMAX, SSUM);
        edge_agg2_kernel<<<E2c, 256, 0, stream>>>(ATT2, c2_src, c2_dst, SSUM, V2, AGGM);

        // gelu -> output projection -> skip gate -> residual -> graph LN
        gelu_kernel<<<g1(OSZ), 256, 0, stream>>>(AGGO, (long long)OSZ);
        gelu_kernel<<<g1(MSZ), 256, 0, stream>>>(AGGM, (long long)MSZ);
        const float* Wo0 = W_out + (l * 2 + 0) * FF; const float* bo0 = b_out + (l * 2 + 0) * Ff;
        const float* Wo1 = W_out + (l * 2 + 1) * FF; const float* bo1 = b_out + (l * 2 + 1) * Ff;
        gemm(stream, AGGO, Ff, 0, Wo0, Ff, 0, bo0, KO, Ff, 0, NOPc, Ff, Ff, 1);
        gemm(stream, AGGM, Ff, 0, Wo1, Ff, 0, bo1, KM, Ff, 0, NMc,  Ff, Ff, 1);

        combine_kernel<<<g1(OSZ), 256, 0, stream>>>(KO, X_OP, skip + l * 2 + 0, VO, (long long)OSZ);
        fill_kernel<<<g1(2), 256, 0, stream>>>(ST, 2, 0.0f);
        stats_kernel<<<512, 256, 0, stream>>>(VO, (long long)OSZ, ST);
        apply_ln_kernel<<<g1(OSZ), 256, 0, stream>>>(VO, X_OP, (long long)OSZ, ln_w + l * Ff, ln_b + l * Ff, ST, 1.0f / (float)OSZ);

        combine_kernel<<<g1(MSZ), 256, 0, stream>>>(KM, X_M, skip + l * 2 + 1, VM, (long long)MSZ);
        fill_kernel<<<g1(2), 256, 0, stream>>>(ST, 2, 0.0f);
        stats_kernel<<<128, 256, 0, stream>>>(VM, (long long)MSZ, ST);
        apply_ln_kernel<<<g1(MSZ), 256, 0, stream>>>(VM, X_M, (long long)MSZ, ln_w + l * Ff, ln_b + l * Ff, ST, 1.0f / (float)MSZ);
    }

    // ---- outputs: fea_j | fea_m | mean_j | mean_m ----
    copy_kernel<<<g1(OSZ), 256, 0, stream>>>(X_OP, d_out, (long long)OSZ);
    copy_kernel<<<g1(MSZ), 256, 0, stream>>>(X_M, d_out + OSZ, (long long)MSZ);
    mean_rows_kernel<<<g1(Bg * Ff), 256, 0, stream>>>(X_OP, d_out + OSZ + MSZ, Nn);
    mean_rows_kernel<<<g1(Bg * Ff), 256, 0, stream>>>(X_M,  d_out + OSZ + MSZ + Bg * Ff, Mm);
}
